// PolyhedronModel_84353157693983
// MI455X (gfx1250) — compile-verified, matrix operand order checked by reading, not run on hardware
//
#include <hip/hip_runtime.h>
#include <hip/hip_bf16.h>

typedef __attribute__((ext_vector_type(16))) _Float16 v16h;
typedef __attribute__((ext_vector_type(8)))  _Float16 v8h;
typedef __attribute__((ext_vector_type(8)))  float    v8f;

#define NN 50000
#define NE 800000
#define NG 512
#define DN 64
#define DE 32
#define ZD 160   // 2*DN + DE
#define NW (DN * ZD)   // 10240 halves per weight matrix

// ---------- helpers ----------
__device__ __forceinline__ v16h cat8(v8h lo, v8h hi) {
    v16h r;
#pragma unroll
    for (int i = 0; i < 8; ++i) { r[i] = lo[i]; r[i + 8] = hi[i]; }
    return r;
}

// Pointer into the virtual z row for edge e at K index k (k is 8-aligned).
__device__ __forceinline__ const float* zptr(const float* __restrict__ x,
                                             const float* __restrict__ ea,
                                             int dst, int src, long e, int k) {
    if (k < 64)  return x + (long)dst * DN + k;
    if (k < 128) return x + (long)src * DN + (k - 64);
    return ea + e * DE + (k - 128);
}

__device__ __forceinline__ void fill8(v16h& a, int base, const float* __restrict__ p) {
    const float4* q = (const float4*)p;
    float4 u = q[0], v = q[1];
    a[base + 0] = (_Float16)u.x; a[base + 1] = (_Float16)u.y;
    a[base + 2] = (_Float16)u.z; a[base + 3] = (_Float16)u.w;
    a[base + 4] = (_Float16)v.x; a[base + 5] = (_Float16)v.y;
    a[base + 6] = (_Float16)v.z; a[base + 7] = (_Float16)v.w;
}

// ---------- prep: W[160][64] f32 -> Wt[64][160] f16 ----------
__global__ void transpose_w_f16(const float* __restrict__ W, _Float16* __restrict__ Wt) {
    int i = blockIdx.x * blockDim.x + threadIdx.x;
    if (i >= NW) return;
    int n = i / ZD, k = i % ZD;
    Wt[i] = (_Float16)W[k * DN + n];
}

__global__ void copy_f32(const float* __restrict__ src, float* __restrict__ dst, int n) {
    int i = blockIdx.x * blockDim.x + threadIdx.x;
    if (i < n) dst[i] = src[i];
}

__global__ void zero_f32(float* __restrict__ dst, int n) {
    int i = blockIdx.x * blockDim.x + threadIdx.x;
    if (i < n) dst[i] = 0.0f;
}

// ---------- CGConv edge kernel: one wave = 16 edges, WMMA f16 GEMM ----------
// Weights (both branches) staged in LDS; B fragments come from ds_load_b128.
__global__ __launch_bounds__(256) void cgconv_edge_wmma(
    const float* __restrict__ xin, const long long* __restrict__ ei,
    const float* __restrict__ ea,
    const _Float16* __restrict__ WtF, const _Float16* __restrict__ WtS,
    const float* __restrict__ bF, const float* __restrict__ bS,
    float* __restrict__ hout)
{
    __shared__ _Float16 sWf[NW];
    __shared__ _Float16 sWs[NW];

    // Cooperative stage: 2 x 20 KB, 16B per transfer, 5 iters/thread/matrix.
    {
        const uint4* gf = (const uint4*)WtF;
        const uint4* gs = (const uint4*)WtS;
        uint4* lf = (uint4*)sWf;
        uint4* ls = (uint4*)sWs;
        for (int i = threadIdx.x; i < NW / 8; i += 256) {   // 1280 uint4 each
            lf[i] = gf[i];
            ls[i] = gs[i];
        }
    }
    __syncthreads();

    const int wave = threadIdx.x >> 5;
    const int lane = threadIdx.x & 31;
    const long tile = (long)blockIdx.x * 8 + wave;
    const long base = tile * 16;
    if (base >= NE) return;

    const int hsel = lane >> 4;     // 0: lanes 0-15, 1: lanes 16-31
    const int m    = lane & 15;     // row for A, column for B/C/D
    const long eM  = base + m;
    const int srcM = (int)ei[eM];        // edge_index[0]
    const int dstM = (int)ei[NE + eM];   // edge_index[1]

    v8f accF[4], accS[4];
    const v8f vzero = {};
#pragma unroll
    for (int t = 0; t < 4; ++t) { accF[t] = vzero; accS[t] = vzero; }

#pragma unroll
    for (int ck = 0; ck < 5; ++ck) {
        const int kb = ck * 32;
        const int k0 = kb + hsel * 8;
        const int k1 = kb + 16 + hsel * 8;

        // Build A fragment for this K-chunk: z row m of the 16-edge tile.
        v16h a;
        fill8(a, 0, zptr(xin, ea, dstM, srcM, eM, k0));
        fill8(a, 8, zptr(xin, ea, dstM, srcM, eM, k1));

        // Stage all 8 B fragments for this chunk from LDS first, ...
        v16h bfrag[8];
#pragma unroll
        for (int nt = 0; nt < 4; ++nt) {
            const int n = nt * 16 + m;
            bfrag[2 * nt + 0] = cat8(*(const v8h*)(sWf + n * ZD + k0),
                                     *(const v8h*)(sWf + n * ZD + k1));
            bfrag[2 * nt + 1] = cat8(*(const v8h*)(sWs + n * ZD + k0),
                                     *(const v8h*)(sWs + n * ZD + k1));
        }
        // ... then run the 8 WMMAs back-to-back.
#pragma unroll
        for (int nt = 0; nt < 4; ++nt) {
            accF[nt] = __builtin_amdgcn_wmma_f32_16x16x32_f16(
                false, a, false, bfrag[2 * nt + 0], (short)0, accF[nt], false, false);
            accS[nt] = __builtin_amdgcn_wmma_f32_16x16x32_f16(
                false, a, false, bfrag[2 * nt + 1], (short)0, accS[nt], false, false);
        }
    }

    // This lane's 8 output rows are edges base + hsel*8 + r.
    int dstR[8];
#pragma unroll
    for (int r = 0; r < 8; ++r) dstR[r] = (int)ei[NE + base + hsel * 8 + r];

#pragma unroll
    for (int nt = 0; nt < 4; ++nt) {
        const int col = nt * 16 + m;
        const float bfv = bF[col];
        const float bsv = bS[col];
        float* colp = hout + col;
#pragma unroll
        for (int r = 0; r < 8; ++r) {
            float f = accF[nt][r] + bfv;
            float s = accS[nt][r] + bsv;
            float sig = 1.0f / (1.0f + __expf(-f));
            float sp  = (s > 0.0f) ? (s + log1pf(__expf(-s))) : log1pf(__expf(s));
            atomicAdd(colp + (long)dstR[r] * DN, sig * sp);
        }
    }
}

// ---------- pooling: segment_sum over batch ----------
__global__ void pool_kernel(const float* __restrict__ h, const long long* __restrict__ batch,
                            float* __restrict__ pooled, int n) {
    int i = blockIdx.x * blockDim.x + threadIdx.x;
    if (i >= n) return;
    int node = i >> 6;      // / DN
    int col  = i & 63;
    int g = (int)batch[node];
    atomicAdd(pooled + (long)g * DN + col, h[i]);
}

// ---------- final linear: out[g] = pooled[g] . W_out + b_out ----------
__global__ void out_kernel(const float* __restrict__ pooled, const float* __restrict__ Wout,
                           const float* __restrict__ bout, float* __restrict__ out) {
    int g = blockIdx.x * blockDim.x + threadIdx.x;
    if (g >= NG) return;
    float acc = bout[0];
#pragma unroll
    for (int c = 0; c < DN; ++c) acc += pooled[(long)g * DN + c] * Wout[c];
    out[g] = acc;
}

extern "C" void kernel_launch(void* const* d_in, const int* in_sizes, int n_in,
                              void* d_out, int out_size, void* d_ws, size_t ws_size,
                              hipStream_t stream) {
    const float*     x    = (const float*)d_in[0];
    const long long* ei   = (const long long*)d_in[1];
    const float*     ea   = (const float*)d_in[2];
    const long long* bat  = (const long long*)d_in[3];
    const float*     Wf1  = (const float*)d_in[4];
    const float*     bf1  = (const float*)d_in[5];
    const float*     Ws1  = (const float*)d_in[6];
    const float*     bs1  = (const float*)d_in[7];
    const float*     Wf2  = (const float*)d_in[8];
    const float*     bf2  = (const float*)d_in[9];
    const float*     Ws2  = (const float*)d_in[10];
    const float*     bs2  = (const float*)d_in[11];
    const float*     Wout = (const float*)d_in[12];
    const float*     bout = (const float*)d_in[13];
    float*           out  = (float*)d_out;

    char* w = (char*)d_ws;
    float*     h1     = (float*)(w);                       // 12.8 MB
    float*     h2     = (float*)(w + 12800000);            // 12.8 MB
    _Float16*  wt     = (_Float16*)(w + 25600000);         // 4 x 10240 halves
    float*     pooled = (float*)(w + 25600000 + 81920);    // 512*64 f32

    transpose_w_f16<<<(NW + 255) / 256, 256, 0, stream>>>(Wf1, wt + 0 * NW);
    transpose_w_f16<<<(NW + 255) / 256, 256, 0, stream>>>(Ws1, wt + 1 * NW);
    transpose_w_f16<<<(NW + 255) / 256, 256, 0, stream>>>(Wf2, wt + 2 * NW);
    transpose_w_f16<<<(NW + 255) / 256, 256, 0, stream>>>(Ws2, wt + 3 * NW);

    const int nh = NN * DN;                 // 3.2M
    copy_f32<<<(nh + 255) / 256, 256, 0, stream>>>(x, h1, nh);
    cgconv_edge_wmma<<<NE / 128, 256, 0, stream>>>(x, ei, ea,
        wt + 0 * NW, wt + 1 * NW, bf1, bs1, h1);

    copy_f32<<<(nh + 255) / 256, 256, 0, stream>>>(h1, h2, nh);
    cgconv_edge_wmma<<<NE / 128, 256, 0, stream>>>(h1, ei, ea,
        wt + 2 * NW, wt + 3 * NW, bf2, bs2, h2);

    zero_f32<<<(NG * DN + 255) / 256, 256, 0, stream>>>(pooled, NG * DN);
    pool_kernel<<<(nh + 255) / 256, 256, 0, stream>>>(h2, bat, pooled, nh);
    out_kernel<<<(NG + 255) / 256, 256, 0, stream>>>(pooled, Wout, bout, out);
}
